// RecurrentStickBreakingConstruction_7859790152215
// MI455X (gfx1250) — matrix operand order repaired and checked by reading, not compiled, not run on hardware
//
#include <hip/hip_runtime.h>

// ---------------------------------------------------------------------------
// Problem: n=256 topics, batch=131072.
//  K1: sequential RNN -> H (255x256), stored f16 (padded to 256x256) in d_ws.
//  K2: logits = z @ H^T via v_wmma_f32_16x16x32_f16 (f16 in, f32 acc),
//      fused sigmoid + stick-breaking scan in LDS, write (131072,256) f32.
// ---------------------------------------------------------------------------

typedef _Float16 v16h __attribute__((ext_vector_type(16)));
typedef float    v8f  __attribute__((ext_vector_type(8)));

#define NTOP 256
#define WSTRIDE 260   // pad 256->260 floats: keeps 16B alignment, spreads LDS banks

// ============================ Kernel 1: RNN ================================
// One workgroup, 1024 threads. Wsum = W_ih + W_hh lives in LDS (f32).
// thread t: output j = t&255, k-quarter part = t>>8 (64 MACs each), LDS reduce.
__global__ __launch_bounds__(1024)
void rnn_seq_kernel(const float* __restrict__ h0,
                    const float* __restrict__ Wih,
                    const float* __restrict__ Whh,
                    const float* __restrict__ bih,
                    const float* __restrict__ bhh,
                    _Float16* __restrict__ Hh)
{
    __shared__ float Ws[NTOP * WSTRIDE];   // 266,240 B (fits 320 KB/WGP LDS)
    __shared__ float hbuf[NTOP];
    __shared__ float psum[4][NTOP];
    __shared__ float bsum[NTOP];

    const int t = threadIdx.x;
    const int j = t & 255;
    const int part = t >> 8;          // 0..3

    // Build combined weights in LDS (coalesced global, conflict-free stores).
    for (int idx = t; idx < NTOP * NTOP; idx += 1024) {
        int jj = idx >> 8, kk = idx & 255;
        Ws[jj * WSTRIDE + kk] = Wih[idx] + Whh[idx];
    }
    if (t < NTOP) bsum[t] = bih[t] + bhh[t];
    if (t < NTOP) Hh[255 * NTOP + t] = (_Float16)0.0f;   // pad row for GEMM N-tile 15

    // Step 0: h1 = tanh(h0 @ W_ih^T + b)   (uses W_ih only, straight from global)
    {
        float p = 0.0f;
        const float* wr = Wih + (size_t)j * NTOP + part * 64;
        const float* hr = h0 + part * 64;
        #pragma unroll
        for (int k = 0; k < 64; k += 4) {
            float4 w = *(const float4*)(wr + k);
            float4 h = *(const float4*)(hr + k);
            p += w.x * h.x + w.y * h.y + w.z * h.z + w.w * h.w;
        }
        psum[part][j] = p;
    }
    __syncthreads();
    if (part == 0) {
        float v = tanhf(psum[0][j] + psum[1][j] + psum[2][j] + psum[3][j] + bsum[j]);
        hbuf[j] = v;
        Hh[j] = (_Float16)v;          // H row 0
    }
    __syncthreads();

    // Steps 1..254: h <- tanh(h @ (W_ih+W_hh)^T + b), all operands in LDS.
    for (int s = 1; s < 255; ++s) {
        float acc = 0.0f;
        const float* wl = &Ws[j * WSTRIDE + part * 64];
        const float* hl = &hbuf[part * 64];
        #pragma unroll
        for (int k = 0; k < 64; k += 4) {
            float4 w = *(const float4*)(wl + k);
            float4 h = *(const float4*)(hl + k);
            acc += w.x * h.x + w.y * h.y + w.z * h.z + w.w * h.w;
        }
        psum[part][j] = acc;
        __syncthreads();
        if (part == 0) {
            float v = tanhf(psum[0][j] + psum[1][j] + psum[2][j] + psum[3][j] + bsum[j]);
            hbuf[j] = v;
            Hh[(size_t)s * NTOP + j] = (_Float16)v;
        }
        __syncthreads();
    }
}

// ==================== Kernel 2: WMMA GEMM + stick-breaking =================
// Block: 256 threads = 8 waves (2 M-waves x 4 N-waves) -> 32x256 logits tile.
// logits = z(131072x256,f32->f16) @ H^T(256x255,f16), f32 accumulate.
__global__ __launch_bounds__(256)
void gemm_sb_kernel(const float* __restrict__ z,
                    const _Float16* __restrict__ Hh,
                    float* __restrict__ out)
{
    __shared__ float Lt[32 * WSTRIDE];   // logits tile -> zc -> output (in place)
    __shared__ float pprod[32][8];

    const int tid   = threadIdx.x;
    const int wave  = tid >> 5;
    const int lane  = tid & 31;
    const int mwave = wave >> 2;         // 0..1
    const int nwave = wave & 3;          // 0..3
    const int lr    = lane & 15;
    const int lh    = lane >> 4;         // half-wave select

    const size_t m = (size_t)blockIdx.x * 32 + mwave * 16 + lr;
    const float* zrow = z + m * NTOP;

    v8f acc[4] = {v8f{0}, v8f{0}, v8f{0}, v8f{0}};

    #pragma unroll
    for (int ks = 0; ks < 8; ++ks) {
        // ---- A fragment: 16x32 f16, per ISA layout (lane: m=lr, K split by lh)
        const int kbA = ks * 32 + lh * 8;
        float4 a0 = *(const float4*)(zrow + kbA);
        float4 a1 = *(const float4*)(zrow + kbA + 4);
        float4 a2 = *(const float4*)(zrow + kbA + 16);
        float4 a3 = *(const float4*)(zrow + kbA + 20);
        v16h a;
        a[0] = (_Float16)a0.x;  a[1] = (_Float16)a0.y;
        a[2] = (_Float16)a0.z;  a[3] = (_Float16)a0.w;
        a[4] = (_Float16)a1.x;  a[5] = (_Float16)a1.y;
        a[6] = (_Float16)a1.z;  a[7] = (_Float16)a1.w;
        a[8] = (_Float16)a2.x;  a[9] = (_Float16)a2.y;
        a[10] = (_Float16)a2.z; a[11] = (_Float16)a2.w;
        a[12] = (_Float16)a3.x; a[13] = (_Float16)a3.y;
        a[14] = (_Float16)a3.z; a[15] = (_Float16)a3.w;

        // ---- B fragments: lane holds halves [8lh,8lh+8) and [16+8lh,24+8lh) of H row n
        const int kbB = ks * 32 + lh * 8;
        #pragma unroll
        for (int nt = 0; nt < 4; ++nt) {
            const int n = nwave * 64 + nt * 16 + lr;   // row 255 is zero pad
            typedef _Float16 v8h_ __attribute__((ext_vector_type(8)));
            const v8h_ b0 = *(const v8h_*)(Hh + (size_t)n * NTOP + kbB);
            const v8h_ b1 = *(const v8h_*)(Hh + (size_t)n * NTOP + kbB + 16);
            v16h b = __builtin_shufflevector(b0, b1, 0,1,2,3,4,5,6,7,8,9,10,11,12,13,14,15);
            acc[nt] = __builtin_amdgcn_wmma_f32_16x16x32_f16(
                false, a, false, b, (short)0, acc[nt], false, false);
        }
    }

    // ---- scatter accumulators to LDS logits tile (m_local, n)
    #pragma unroll
    for (int nt = 0; nt < 4; ++nt) {
        const int n = nwave * 64 + nt * 16 + lr;
        #pragma unroll
        for (int r = 0; r < 8; ++r) {
            const int ml = mwave * 16 + lh * 8 + r;
            Lt[ml * WSTRIDE + n] = acc[nt][r];
        }
    }
    __syncthreads();

    // ---- stick-breaking: 8 threads per row, 32-element segments
    const int row = tid >> 3;
    const int seg = tid & 7;
    float* lrow = &Lt[row * WSTRIDE];
    const int n0 = seg * 32;

    float p = 1.0f;
    #pragma unroll 4
    for (int i = 0; i < 32; ++i) {
        const int n = n0 + i;
        if (n < 255) {
            const float l   = lrow[n];
            const float eta = 1.0f / (1.0f + expf(-l));           // sigmoid(logit)
            const float x   = eta - logf((float)(255 - n));        // - log(offset)
            float zc = 1.0f / (1.0f + expf(-x));
            zc = fminf(fmaxf(zc, 1.175494351e-38f), 0.99999988079071045f);
            lrow[n] = zc;
            p *= (1.0f - zc);
        }
    }
    pprod[row][seg] = p;
    __syncthreads();

    float run = 1.0f;                       // exclusive prefix of segment products
    for (int s2 = 0; s2 < seg; ++s2) run *= pprod[row][s2];
    #pragma unroll 4
    for (int i = 0; i < 32; ++i) {
        const int n = n0 + i;
        if (n < 255) {
            const float zc = lrow[n];
            lrow[n] = zc * run;             // out[n] = zc[n] * prod_{j<n}(1-zc[j])
            run *= (1.0f - zc);
        } else {
            lrow[255] = run;                // out[255] = full cumprod
        }
    }
    __syncthreads();

    // ---- coalesced 128-bit store of the 32x256 tile
    float* orow = out + (size_t)blockIdx.x * 32 * NTOP;
    #pragma unroll
    for (int v = 0; v < 8; ++v) {
        const int idx = v * 256 + tid;      // 2048 float4s
        const int r   = idx >> 6;
        const int c4  = (idx & 63) << 2;
        *(float4*)(orow + r * NTOP + c4) = *(const float4*)(&Lt[r * WSTRIDE + c4]);
    }
}

// ============================= launcher ====================================
extern "C" void kernel_launch(void* const* d_in, const int* in_sizes, int n_in,
                              void* d_out, int out_size, void* d_ws, size_t ws_size,
                              hipStream_t stream) {
    const float* z    = (const float*)d_in[0];   // (131072, 256)
    const float* h0   = (const float*)d_in[1];   // (1, 256)
    const float* Wih  = (const float*)d_in[2];   // (256, 256)
    const float* Whh  = (const float*)d_in[3];   // (256, 256)
    const float* bih  = (const float*)d_in[4];   // (256,)
    const float* bhh  = (const float*)d_in[5];   // (256,)
    float* out = (float*)d_out;                  // (131072, 256)

    _Float16* Hh = (_Float16*)d_ws;              // 256x256 f16 (row 255 zero pad)

    rnn_seq_kernel<<<1, 1024, 0, stream>>>(h0, Wih, Whh, bih, bhh, Hh);

    const int nblocks = 131072 / 32;             // 4096 blocks of 32 rows
    gemm_sb_kernel<<<nblocks, 256, 0, stream>>>(z, Hh, out);
}
